// GnBlock_45509473468797
// MI455X (gfx1250) — compile-verified, matrix operand order checked by reading, not run on hardware
//
#include <hip/hip_runtime.h>
#include <hip/hip_bf16.h>

#define NODES 100000
#define EDGES 1600000
#define DIM   64

typedef __attribute__((ext_vector_type(16))) __bf16 bf16x16;
typedef __attribute__((ext_vector_type(8)))  __bf16 bf16x8;
typedef __attribute__((ext_vector_type(8)))  float  f32x8;

// ---------------- WMMA fragment helpers (wave32, 16x16x32 bf16) -------------
// A (16x32, 16-bit): lanes 0-15 rows M=0..15 with K=0..7,16..23 ; lanes 16-31
// same rows with K=8..15,24..31.  Element j -> K = (j&7) + hi*8 + (j>>3)*16.
__device__ __forceinline__ bf16x16 load_a_frag_g(const float* __restrict__ arow,
                                                 int kbase, int hi) {
  const float4* p0 = (const float4*)(arow + kbase + hi * 8);
  const float4* p1 = (const float4*)(arow + kbase + 16 + hi * 8);
  float4 r0 = p0[0], r1 = p0[1], r2 = p1[0], r3 = p1[1];
  bf16x16 a;
  a[0]  = (__bf16)r0.x; a[1]  = (__bf16)r0.y; a[2]  = (__bf16)r0.z; a[3]  = (__bf16)r0.w;
  a[4]  = (__bf16)r1.x; a[5]  = (__bf16)r1.y; a[6]  = (__bf16)r1.z; a[7]  = (__bf16)r1.w;
  a[8]  = (__bf16)r2.x; a[9]  = (__bf16)r2.y; a[10] = (__bf16)r2.z; a[11] = (__bf16)r2.w;
  a[12] = (__bf16)r3.x; a[13] = (__bf16)r3.y; a[14] = (__bf16)r3.z; a[15] = (__bf16)r3.w;
  return a;
}

// Fragment from an LDS row where K is contiguous (used for B cols and LDS A rows)
__device__ __forceinline__ bf16x16 load_frag_lds(const __bf16* row, int kbase, int hi) {
  bf16x8 lo = *(const bf16x8*)(row + kbase + hi * 8);
  bf16x8 hv = *(const bf16x8*)(row + kbase + 16 + hi * 8);
  bf16x16 b;
#pragma unroll
  for (int j = 0; j < 8; ++j) { b[j] = lo[j]; b[j + 8] = hv[j]; }
  return b;
}

__device__ __forceinline__ f32x8 wmma2(bf16x16 a0, bf16x16 b0, bf16x16 a1, bf16x16 b1,
                                       f32x8 c) {
  c = __builtin_amdgcn_wmma_f32_16x16x32_bf16(false, a0, false, b0, (short)0, c,
                                              false, false);
  c = __builtin_amdgcn_wmma_f32_16x16x32_bf16(false, a1, false, b1, (short)0, c,
                                              false, false);
  return c;
}

// ------------------ fused N x 64 @ 64 x 64 GEMM (ChebConv terms) -------------
// block = 256 threads = 8 waves; each wave one 16x16 tile; block = 32 rows.
template <bool ACCUM, bool BIAS>
__global__ __launch_bounds__(256) void gemm_nd(const float* __restrict__ A,
                                               const float* __restrict__ W,
                                               const float* __restrict__ bias,
                                               float* __restrict__ Out,
                                               int nrows) {
  __shared__ __attribute__((aligned(16))) __bf16 Wt[DIM * DIM];
  const int tid = threadIdx.x;
  for (int i = tid; i < DIM * DIM; i += 256) {
    int r = i >> 6, c = i & 63;
    Wt[c * DIM + r] = (__bf16)W[i];   // transpose: K becomes contiguous
  }
  __syncthreads();

  const int wave = tid >> 5, lane = tid & 31;
  const int row0 = (blockIdx.x * 2 + (wave >> 2)) * 16;
  const int col0 = (wave & 3) * 16;
  if (row0 >= nrows) return;                  // wave-uniform; EXEC stays full
  const int n = lane & 15, hi = lane >> 4;

  const float* arow = A + (size_t)(row0 + n) * DIM;
  bf16x16 a0 = load_a_frag_g(arow, 0, hi);
  bf16x16 a1 = load_a_frag_g(arow, 32, hi);
  const __bf16* wcol = Wt + (col0 + n) * DIM;
  bf16x16 b0 = load_frag_lds(wcol, 0, hi);
  bf16x16 b1 = load_frag_lds(wcol, 32, hi);

  float* orow = Out + (size_t)(row0 + hi * 8) * DIM + col0 + n;
  f32x8 c = {};
  if (ACCUM) {
#pragma unroll
    for (int r = 0; r < 8; ++r) c[r] = orow[r * DIM];
  }
  c = wmma2(a0, b0, a1, b1, c);

  const float bv = BIAS ? bias[col0 + n] : 0.0f;
#pragma unroll
  for (int r = 0; r < 8; ++r) {
    float v = c[r];
    if (BIAS) v += bv;
    orow[r * DIM] = v;
  }
}

// ------------------ fused MLP (4 layers) + LayerNorm -------------------------
// One block = 32 rows x 64 cols; all four weights staged transposed in LDS as
// bf16; activations ping-pong through LDS; LN reduces in LDS at the end.
__global__ __launch_bounds__(256) void mlp_ln_fused(
    const float* In,                                  // may alias Out
    const float* __restrict__ W1, const float* __restrict__ B1,
    const float* __restrict__ W2, const float* __restrict__ B2,
    const float* __restrict__ W3, const float* __restrict__ B3,
    const float* __restrict__ W4, const float* __restrict__ B4,
    const float* __restrict__ g, const float* __restrict__ bln,
    float* Out, int nrows) {
  __shared__ __attribute__((aligned(16))) __bf16 Wt[4][DIM * DIM];  // 32 KB
  __shared__ __attribute__((aligned(16))) __bf16 hA[32 * DIM];      // 4 KB
  __shared__ __attribute__((aligned(16))) __bf16 hB[32 * DIM];      // 4 KB
  __shared__ __attribute__((aligned(16))) float  h4[32 * DIM];      // 8 KB

  const int tid = threadIdx.x;
  const float* Ws[4] = {W1, W2, W3, W4};
#pragma unroll
  for (int l = 0; l < 4; ++l)
    for (int i = tid; i < DIM * DIM; i += 256) {
      int r = i >> 6, c = i & 63;
      Wt[l][c * DIM + r] = (__bf16)Ws[l][i];
    }
  __syncthreads();

  const int wave = tid >> 5, lane = tid & 31;
  const int rowL = (wave >> 2) * 16;           // local row tile: 0 or 16
  const int col0 = (wave & 3) * 16;
  const int n = lane & 15, hi = lane >> 4;
  const int rowbase = blockIdx.x * 32;
  if (rowbase >= nrows) return;

  const float bv1 = B1[col0 + n], bv2 = B2[col0 + n];
  const float bv3 = B3[col0 + n], bv4 = B4[col0 + n];

  // ---- layer 1: A from global ----
  {
    const float* arow = In + (size_t)(rowbase + rowL + n) * DIM;
    bf16x16 a0 = load_a_frag_g(arow, 0, hi);
    bf16x16 a1 = load_a_frag_g(arow, 32, hi);
    const __bf16* wc = &Wt[0][(col0 + n) * DIM];
    f32x8 c = wmma2(a0, load_frag_lds(wc, 0, hi), a1, load_frag_lds(wc, 32, hi),
                    f32x8{});
    __bf16* hrow = hA + (rowL + hi * 8) * DIM + col0 + n;
#pragma unroll
    for (int r = 0; r < 8; ++r) hrow[r * DIM] = (__bf16)fmaxf(c[r] + bv1, 0.0f);
  }
  __syncthreads();

  // ---- layer 2: hA -> hB ----
  {
    const __bf16* ar = hA + (rowL + n) * DIM;
    const __bf16* wc = &Wt[1][(col0 + n) * DIM];
    f32x8 c = wmma2(load_frag_lds(ar, 0, hi), load_frag_lds(wc, 0, hi),
                    load_frag_lds(ar, 32, hi), load_frag_lds(wc, 32, hi), f32x8{});
    __bf16* hrow = hB + (rowL + hi * 8) * DIM + col0 + n;
#pragma unroll
    for (int r = 0; r < 8; ++r) hrow[r * DIM] = (__bf16)fmaxf(c[r] + bv2, 0.0f);
  }
  __syncthreads();

  // ---- layer 3: hB -> hA ----
  {
    const __bf16* ar = hB + (rowL + n) * DIM;
    const __bf16* wc = &Wt[2][(col0 + n) * DIM];
    f32x8 c = wmma2(load_frag_lds(ar, 0, hi), load_frag_lds(wc, 0, hi),
                    load_frag_lds(ar, 32, hi), load_frag_lds(wc, 32, hi), f32x8{});
    __bf16* hrow = hA + (rowL + hi * 8) * DIM + col0 + n;
#pragma unroll
    for (int r = 0; r < 8; ++r) hrow[r * DIM] = (__bf16)fmaxf(c[r] + bv3, 0.0f);
  }
  __syncthreads();

  // ---- layer 4 (no ReLU): hA -> h4 (f32) ----
  {
    const __bf16* ar = hA + (rowL + n) * DIM;
    const __bf16* wc = &Wt[3][(col0 + n) * DIM];
    f32x8 c = wmma2(load_frag_lds(ar, 0, hi), load_frag_lds(wc, 0, hi),
                    load_frag_lds(ar, 32, hi), load_frag_lds(wc, 32, hi), f32x8{});
    float* hrow = h4 + (rowL + hi * 8) * DIM + col0 + n;
#pragma unroll
    for (int r = 0; r < 8; ++r) hrow[r * DIM] = c[r] + bv4;
  }
  __syncthreads();

  // ---- LayerNorm: each wave handles 4 of the 32 rows ----
  const float2 gg = ((const float2*)g)[lane];
  const float2 bb = ((const float2*)bln)[lane];
#pragma unroll
  for (int rr = 0; rr < 4; ++rr) {
    int rl = wave * 4 + rr;
    float2 v = *(const float2*)&h4[rl * DIM + lane * 2];
    float s = v.x + v.y;
#pragma unroll
    for (int off = 16; off > 0; off >>= 1) s += __shfl_xor(s, off, 32);
    float mu = s * (1.0f / 64.0f);
    float dx = v.x - mu, dy = v.y - mu;
    float q = dx * dx + dy * dy;
#pragma unroll
    for (int off = 16; off > 0; off >>= 1) q += __shfl_xor(q, off, 32);
    float inv = rsqrtf(q * (1.0f / 64.0f) + 1e-5f);
    float2 o;
    o.x = dx * inv * gg.x + bb.x;
    o.y = dy * inv * gg.y + bb.y;
    *(float2*)(Out + (size_t)(rowbase + rl) * DIM + lane * 2) = o;
  }
}

// ------------------ graph preprocessing: degrees + CSR(dst) ------------------
__global__ void edge_pass1(const int* __restrict__ src, const int* __restrict__ dst,
                           float* __restrict__ deg, int* __restrict__ counts, int e) {
  int t = blockIdx.x * 256 + threadIdx.x;
  if (t < e) {
    int s = src[t], d = dst[t];
    if (s != d) {
      atomicAdd(deg + s, 1.0f);   // out-degree over src (valid edges)
      atomicAdd(counts + d, 1);   // CSR row sizes by dst
    }
  }
}

__global__ void deg_rsqrt(float* __restrict__ deg, int nn) {
  int t = blockIdx.x * 256 + threadIdx.x;
  if (t < nn) {
    float d = deg[t];
    deg[t] = d > 0.0f ? rsqrtf(d) : 0.0f;
  }
}

// two-level exclusive scan over counts -> row_start
__global__ void scan_block(const int* __restrict__ in, int* __restrict__ out,
                           int* __restrict__ bsum, int nn) {
  __shared__ int s[256];
  int t = threadIdx.x, gidx = blockIdx.x * 256 + t;
  int v = (gidx < nn) ? in[gidx] : 0;
  s[t] = v;
  __syncthreads();
  for (int off = 1; off < 256; off <<= 1) {
    int add = (t >= off) ? s[t - off] : 0;
    __syncthreads();
    s[t] += add;
    __syncthreads();
  }
  if (gidx < nn) out[gidx] = s[t] - v;          // exclusive
  if (t == 255) bsum[blockIdx.x] = s[255];
}

__global__ void scan_single(int* __restrict__ data, int nn) {  // nn <= 512
  __shared__ int s[512];
  int t = threadIdx.x;
  int v = (t < nn) ? data[t] : 0;
  s[t] = v;
  __syncthreads();
  for (int off = 1; off < 512; off <<= 1) {
    int add = (t >= off) ? s[t - off] : 0;
    __syncthreads();
    s[t] += add;
    __syncthreads();
  }
  if (t < nn) data[t] = s[t] - v;
}

__global__ void add_off(int* __restrict__ row_start, const int* __restrict__ bsum,
                        int* __restrict__ cursor, int nn) {
  int t = blockIdx.x * 256 + threadIdx.x;
  if (t < nn) {
    int v = row_start[t] + bsum[blockIdx.x];
    row_start[t] = v;
    cursor[t] = v;
  }
}

__global__ void csr_fill(const int* __restrict__ src, const int* __restrict__ dst,
                         const float* __restrict__ isq, int* __restrict__ cursor,
                         int* __restrict__ csr_src, float* __restrict__ csr_w, int e) {
  int t = blockIdx.x * 256 + threadIdx.x;
  if (t < e) {
    int s = src[t], d = dst[t];
    if (s != d) {
      int pos = atomicAdd(cursor + d, 1);
      csr_src[pos] = s;
      csr_w[pos] = -isq[s] * isq[d];
    }
  }
}

// vout[node] = scale * sum_{e in CSR row} w_e * vin[src_e]  (- sub[node] if set)
// 64 threads per node; gathers hit L2 (feature matrix is L2-resident); NO atomics.
__global__ void prop_gather(const float* __restrict__ vin, const int* __restrict__ rs,
                            const int* __restrict__ cnt, const int* __restrict__ csrc,
                            const float* __restrict__ cw, const float* __restrict__ sub,
                            float scale, float* __restrict__ vout, int nnodes) {
  int t = blockIdx.x * 256 + threadIdx.x;
  int node = t >> 6, f = t & 63;
  if (node >= nnodes) return;
  int s0 = rs[node], c = cnt[node];
  float acc = 0.0f;
  for (int j = s0; j < s0 + c; ++j)
    acc += cw[j] * vin[(size_t)csrc[j] * DIM + f];
  float v = scale * acc;
  if (sub) v -= sub[(size_t)node * DIM + f];
  vout[(size_t)node * DIM + f] = v;
}

// ---------------------------------------------------------------------------
extern "C" void kernel_launch(void* const* d_in, const int* in_sizes, int n_in,
                              void* d_out, int out_size, void* d_ws, size_t ws_size,
                              hipStream_t stream) {
  const float* x   = (const float*)d_in[0];
  const int*   ei  = (const int*)d_in[1];
  const int*   src = ei;
  const int*   dst = ei + EDGES;
  const float* cw  = (const float*)d_in[2];   // [5][64][64]
  const float* cb  = (const float*)d_in[3];
  const float* w1  = (const float*)d_in[4];
  const float* b1  = (const float*)d_in[5];
  const float* w2  = (const float*)d_in[6];
  const float* b2  = (const float*)d_in[7];
  const float* w3  = (const float*)d_in[8];
  const float* b3  = (const float*)d_in[9];
  const float* w4  = (const float*)d_in[10];
  const float* b4  = (const float*)d_in[11];
  const float* lng = (const float*)d_in[12];
  const float* lnb = (const float*)d_in[13];
  float* out = (float*)d_out;

  const size_t ND = (size_t)NODES * DIM;
  // workspace layout (~91 MB)
  float* deg   = (float*)d_ws;          // N
  float* csr_w = deg + NODES;           // E
  float* bufA  = csr_w + EDGES;         // N*64
  float* bufB  = bufA + ND;             // N*64
  float* bufC  = bufB + ND;             // N*64
  int* counts    = (int*)(bufC + ND);   // N
  int* row_start = counts + NODES;      // N
  int* cursor    = row_start + NODES;   // N
  int* bsum      = cursor + NODES;      // 512
  int* csr_src   = bsum + 512;          // E

  const int EB  = (EDGES + 255) / 256;
  const int NB  = (NODES + 255) / 256;   // 391
  const int PG  = (NODES * 64) / 256;    // 25000 (exact)
  const int GG  = (NODES + 31) / 32;     // 3125 (exact)

  // --- degrees + CSR-by-dst build (once; amortized over 4 propagations) ---
  hipMemsetAsync(deg, 0, NODES * sizeof(float), stream);
  hipMemsetAsync(counts, 0, NODES * sizeof(int), stream);
  edge_pass1<<<EB, 256, 0, stream>>>(src, dst, deg, counts, EDGES);
  deg_rsqrt<<<NB, 256, 0, stream>>>(deg, NODES);
  scan_block<<<NB, 256, 0, stream>>>(counts, row_start, bsum, NODES);
  scan_single<<<1, 512, 0, stream>>>(bsum, NB);
  add_off<<<NB, 256, 0, stream>>>(row_start, bsum, cursor, NODES);
  csr_fill<<<EB, 256, 0, stream>>>(src, dst, deg, cursor, csr_src, csr_w, EDGES);

  // --- ChebConv: out accumulates sum_k T_k(L) x @ W_k  (atomic-free props) ---
  gemm_nd<false, false><<<GG, 256, 0, stream>>>(x, cw, nullptr, out, NODES);

  prop_gather<<<PG, 256, 0, stream>>>(x, row_start, counts, csr_src, csr_w,
                                      nullptr, 1.0f, bufA, NODES);           // tx1
  gemm_nd<true, false><<<GG, 256, 0, stream>>>(bufA, cw + 4096, nullptr, out, NODES);

  prop_gather<<<PG, 256, 0, stream>>>(bufA, row_start, counts, csr_src, csr_w,
                                      x, 2.0f, bufB, NODES);                 // tx2
  gemm_nd<true, false><<<GG, 256, 0, stream>>>(bufB, cw + 2 * 4096, nullptr, out, NODES);

  prop_gather<<<PG, 256, 0, stream>>>(bufB, row_start, counts, csr_src, csr_w,
                                      bufA, 2.0f, bufC, NODES);              // tx3
  gemm_nd<true, false><<<GG, 256, 0, stream>>>(bufC, cw + 3 * 4096, nullptr, out, NODES);

  prop_gather<<<PG, 256, 0, stream>>>(bufC, row_start, counts, csr_src, csr_w,
                                      bufB, 2.0f, bufA, NODES);              // tx4
  gemm_nd<true, true><<<GG, 256, 0, stream>>>(bufA, cw + 4 * 4096, cb, out, NODES);

  // --- fused MLP (ReLU x3 -> Linear) + LayerNorm, in-place on d_out ---
  mlp_ln_fused<<<GG, 256, 0, stream>>>(out, w1, b1, w2, b2, w3, b3, w4, b4,
                                       lng, lnb, out, NODES);
}
